// PerformerAttention_76355928588534
// MI455X (gfx1250) — matrix operand order repaired and checked by reading, not compiled, not run on hardware
//
#include <hip/hip_runtime.h>
#include <hip/hip_bf16.h>
#include <math.h>

typedef __attribute__((ext_vector_type(16))) __bf16 v16bf;
typedef __attribute__((ext_vector_type(8)))  __bf16 v8bf;
typedef __attribute__((ext_vector_type(4)))  __bf16 v4bf;
typedef __attribute__((ext_vector_type(8)))  float  v8f;

namespace cfg {
constexpr int B   = 4;
constexpr int T   = 4096;
constexpr int DIM = 1024;
constexpr int M   = 512;
constexpr float EPS = 1e-8f;
constexpr int BK  = 32;   // K-step per WMMA pass
constexpr int BKP = 48;   // padded LDS row (bf16 elems): 96B rows, 16B-aligned
}

#define WMMA_BF16(a, b, c) \
  __builtin_amdgcn_wmma_f32_16x16x32_bf16(false, (a), false, (b), (short)0, (c), false, false)

// ---------------------------------------------------------------------------
// gfx1250 async global->LDS copy (ASYNCcnt-tracked), GV addressing mode:
// per-lane 32-bit LDS dest address + 64-bit global address VGPR pair.
// ---------------------------------------------------------------------------
__device__ inline void async_copy_b128(uint32_t lds_addr, const void* gptr) {
  asm volatile("global_load_async_to_lds_b128 %0, %1, off"
               :: "v"(lds_addr), "v"((uint64_t)(uintptr_t)gptr) : "memory");
}
__device__ inline void wait_async_zero() {
  asm volatile("s_wait_asynccnt 0x0" ::: "memory");
}

// ---------------------------------------------------------------------------
// WMMA fragment loaders (CDNA5 wave32 layouts, bf16 A 16x32 / B 32x16).
// LDS tile layout for both: [16 rows][BKP], row = M-row (A) or N-column (B),
// columns = K (contiguous).
// ---------------------------------------------------------------------------
__device__ inline v16bf load_frag_a(const __bf16* s) {
  using namespace cfg;
  const int lane = threadIdx.x & 31;
  const int r = lane & 15, h = lane >> 4;
  const __bf16* row = s + r * BKP;
  v16bf a;
#pragma unroll
  for (int v = 0; v < 8; ++v) {
    // lane<16: K = 0..7 (vgpr0-3), 16..23 (vgpr4-7); lane>=16: +8
    const int k = ((v < 4) ? (2 * v) : (2 * v + 8)) + 8 * h;
    a[2 * v]     = row[k];
    a[2 * v + 1] = row[k + 1];
  }
  return a;
}

__device__ inline v16bf load_frag_b(const __bf16* s) {
  using namespace cfg;
  const int lane = threadIdx.x & 31;
  const int n = lane & 15, h = lane >> 4;
  const __bf16* row = s + n * BKP;
  v16bf b;
#pragma unroll
  for (int v = 0; v < 8; ++v) {
    // lane<16: K = 0..7 (vgpr0-3), 16..23 (vgpr4-7); lane>=16: +8
    const int k = ((v < 4) ? (2 * v) : (2 * v + 8)) + 8 * h;
    b[2 * v]     = row[k];
    b[2 * v + 1] = row[k + 1];
  }
  return b;
}

// ---------------------------------------------------------------------------
// xd[b,t] = 0.5 * || mask * x ||^2
// ---------------------------------------------------------------------------
__global__ __launch_bounds__(256)
void rownorm_kernel(const float* __restrict__ X, const float* __restrict__ mask,
                    float* __restrict__ xd, int use_mask) {
  using namespace cfg;
  const int row  = blockIdx.x * 8 + (threadIdx.x >> 5);
  const int lane = threadIdx.x & 31;
  if (row >= B * T) return;
  const float* x = X + (size_t)row * DIM;
  float s = 0.f;
  for (int i = lane; i < DIM; i += 32) { float v = x[i]; s += v * v; }
  if (use_mask) { float mv = mask[row]; s *= mv * mv; }
#pragma unroll
  for (int off = 16; off; off >>= 1) s += __shfl_xor(s, off, 32);
  if (lane == 0) xd[row] = 0.5f * s;
}

// ---------------------------------------------------------------------------
// Vb = bf16(V * mask)
// ---------------------------------------------------------------------------
__global__ __launch_bounds__(256)
void convv_kernel(const float* __restrict__ V, const float* __restrict__ mask,
                  __bf16* __restrict__ Vb) {
  using namespace cfg;
  const size_t n4 = (size_t)B * T * DIM / 4;
  for (size_t idx = (size_t)blockIdx.x * blockDim.x + threadIdx.x; idx < n4;
       idx += (size_t)gridDim.x * blockDim.x) {
    float4 v = reinterpret_cast<const float4*>(V)[idx];
    float mv = mask[(idx * 4) / DIM];
    v4bf o;
    o[0] = (__bf16)(v.x * mv); o[1] = (__bf16)(v.y * mv);
    o[2] = (__bf16)(v.z * mv); o[3] = (__bf16)(v.w * mv);
    reinterpret_cast<v4bf*>(Vb)[idx] = o;
  }
}

// ---------------------------------------------------------------------------
// P[b,t,m] = exp( (mask*X) . w[m] - xd[b,t] ) / sqrt(M), bf16 out.
// 64(t) x 128(m) block tile, 4 waves; wave = 32x64 = 2x4 16x16 frags.
// ---------------------------------------------------------------------------
template <bool USE_MASK>
__global__ __launch_bounds__(128)
void featurize_kernel(const float* __restrict__ X, const float* __restrict__ w,
                      const float* __restrict__ xd, const float* __restrict__ mask,
                      __bf16* __restrict__ P, float rscale) {
  using namespace cfg;
  __shared__ __align__(16) __bf16 As[64 * BKP];    // [t-row][k]
  __shared__ __align__(16) __bf16 Bs[128 * BKP];   // [m-col][k]
  const int b  = blockIdx.z;
  const int t0 = blockIdx.x * 64;
  const int m0 = blockIdx.y * 128;
  const float* Xb    = X    + (size_t)b * T * DIM;
  const float* maskb = mask + (size_t)b * T;
  const float* xdb   = xd   + (size_t)b * T;
  __bf16*      Pb    = P    + (size_t)b * T * M;

  const int tid  = threadIdx.x;
  const int lane = tid & 31;
  const int wave = tid >> 5;
  const int wr = wave >> 1, wc = wave & 1;
  const int rbase = tid >> 3;       // 0..15 (fill row base)
  const int q     = tid & 7;        // float4 column within 32-wide K tile

  // Hoist mask values (k-invariant) out of the K loop.
  float am[4];
#pragma unroll
  for (int i = 0; i < 4; ++i)
    am[i] = USE_MASK ? maskb[t0 + rbase + 16 * i] : 1.0f;

  v8f acc0[4] = {}, acc1[4] = {};

  for (int k0 = 0; k0 < DIM; k0 += BK) {
    __syncthreads();
#pragma unroll
    for (int i = 0; i < 4; ++i) {               // A: 64 rows x 8 float4
      const int r = rbase + 16 * i;
      float4 v = *reinterpret_cast<const float4*>(Xb + (size_t)(t0 + r) * DIM + k0 + q * 4);
      const float mv = am[i];
      __bf16* da = &As[r * BKP + q * 4];
      da[0] = (__bf16)(v.x * mv); da[1] = (__bf16)(v.y * mv);
      da[2] = (__bf16)(v.z * mv); da[3] = (__bf16)(v.w * mv);
    }
#pragma unroll
    for (int i = 0; i < 8; ++i) {               // B: 128 rows x 8 float4
      const int r = rbase + 16 * i;
      float4 u = *reinterpret_cast<const float4*>(w + (size_t)(m0 + r) * DIM + k0 + q * 4);
      __bf16* db = &Bs[r * BKP + q * 4];
      db[0] = (__bf16)u.x; db[1] = (__bf16)u.y;
      db[2] = (__bf16)u.z; db[3] = (__bf16)u.w;
    }
    __syncthreads();
    if (k0 + BK < DIM) {                        // warm next K tile in caches
      __builtin_prefetch(Xb + (size_t)(t0 + (tid >> 1)) * DIM + k0 + BK, 0, 0);
      __builtin_prefetch(w  + (size_t)(m0 + tid) * DIM + k0 + BK, 0, 0);
    }
    v16bf a0 = load_frag_a(&As[(wr * 32 +  0) * BKP]);
    v16bf a1 = load_frag_a(&As[(wr * 32 + 16) * BKP]);
#pragma unroll
    for (int j = 0; j < 4; ++j) {
      v16bf bj = load_frag_b(&Bs[(wc * 64 + j * 16) * BKP]);
      acc0[j] = WMMA_BF16(a0, bj, acc0[j]);
      acc1[j] = WMMA_BF16(a1, bj, acc1[j]);
    }
  }

  const int col = lane & 15, h = lane >> 4;
#pragma unroll
  for (int s = 0; s < 2; ++s)
#pragma unroll
    for (int j = 0; j < 4; ++j) {
      v8f c = s ? acc1[j] : acc0[j];
#pragma unroll
      for (int v = 0; v < 8; ++v) {
        const int trow = t0 + wr * 32 + s * 16 + v + 8 * h;
        const int mcol = m0 + wc * 64 + j * 16 + col;
        const float p = __expf(c[v] - xdb[trow]) * rscale;
        Pb[(size_t)trow * M + mcol] = (__bf16)p;
      }
    }
}

// ---------------------------------------------------------------------------
// kpsum[b,m] = sum_t kp[b,t,m]
// ---------------------------------------------------------------------------
__global__ __launch_bounds__(256)
void kpsum_kernel(const __bf16* __restrict__ kp, float* __restrict__ kpsum) {
  using namespace cfg;
  const int b = blockIdx.y;
  const int m = blockIdx.x * 256 + threadIdx.x;
  const __bf16* kpb = kp + (size_t)b * T * M;
  float s = 0.f;
  for (int t = 0; t < T; ++t) s += (float)kpb[(size_t)t * M + m];
  kpsum[b * M + m] = s;
}

// ---------------------------------------------------------------------------
// dscale[b,t] = 1 / (qp[b,t,:].kpsum[b,:] + EPS)^2   (y divided by D twice)
// ---------------------------------------------------------------------------
__global__ __launch_bounds__(256)
void dscale_kernel(const __bf16* __restrict__ qp, const float* __restrict__ kpsum,
                   float* __restrict__ dscale) {
  using namespace cfg;
  const int row  = blockIdx.x * 8 + (threadIdx.x >> 5);
  const int lane = threadIdx.x & 31;
  if (row >= B * T) return;
  const int b = row / T;
  const __bf16* q  = qp + (size_t)row * M;
  const float*  ks = kpsum + b * M;
  float s = 0.f;
  for (int i = lane; i < M; i += 32) s += (float)q[i] * ks[i];
#pragma unroll
  for (int off = 16; off; off >>= 1) s += __shfl_xor(s, off, 32);
  if (lane == 0) { const float d = s + EPS; dscale[row] = 1.0f / (d * d); }
}

// ---------------------------------------------------------------------------
// kptv[b,n,m] = sum_t Vb[b,t,n] * kp[b,t,m]   (contraction over T; both
// operands t-major -> write-side transpose through LDS).
// 64(n) x 128(m) block tile, 4 waves; wave = 32x64 = 2x4 frags.
// ---------------------------------------------------------------------------
__global__ __launch_bounds__(128)
void kptv_kernel(const __bf16* __restrict__ Vb, const __bf16* __restrict__ kp,
                 __bf16* __restrict__ kptvb) {
  using namespace cfg;
  __shared__ __align__(16) __bf16 As[64 * BKP];    // [n][t]
  __shared__ __align__(16) __bf16 Bs[128 * BKP];   // [m][t]
  const int b  = blockIdx.z;
  const int n0 = blockIdx.x * 64;
  const int m0 = blockIdx.y * 128;
  const __bf16* Vbb = Vb + (size_t)b * T * DIM;
  const __bf16* kpb = kp + (size_t)b * T * M;
  __bf16*       Cb  = kptvb + (size_t)b * DIM * M;

  const int tid  = threadIdx.x;
  const int lane = tid & 31;
  const int wave = tid >> 5;
  const int wr = wave >> 1, wc = wave & 1;
  const int tloc = tid >> 2;          // 0..31 (k-row of source)
  const int cbA  = (tid & 3) * 16;    // 16 n-cols per thread
  const int cbB  = (tid & 3) * 32;    // 32 m-cols per thread

  v8f acc0[4] = {}, acc1[4] = {};

  for (int t0 = 0; t0 < T; t0 += BK) {
    __syncthreads();
    {
      const __bf16* srcA = Vbb + (size_t)(t0 + tloc) * DIM + n0 + cbA;
      v8bf a0v = *reinterpret_cast<const v8bf*>(srcA);
      v8bf a1v = *reinterpret_cast<const v8bf*>(srcA + 8);
      const __bf16* srcB = kpb + (size_t)(t0 + tloc) * M + m0 + cbB;
      v8bf b0v = *reinterpret_cast<const v8bf*>(srcB);
      v8bf b1v = *reinterpret_cast<const v8bf*>(srcB + 8);
      v8bf b2v = *reinterpret_cast<const v8bf*>(srcB + 16);
      v8bf b3v = *reinterpret_cast<const v8bf*>(srcB + 24);
#pragma unroll
      for (int j = 0; j < 8; ++j) {
        As[(cbA + j)      * BKP + tloc] = a0v[j];
        As[(cbA + 8 + j)  * BKP + tloc] = a1v[j];
        Bs[(cbB + j)      * BKP + tloc] = b0v[j];
        Bs[(cbB + 8 + j)  * BKP + tloc] = b1v[j];
        Bs[(cbB + 16 + j) * BKP + tloc] = b2v[j];
        Bs[(cbB + 24 + j) * BKP + tloc] = b3v[j];
      }
    }
    __syncthreads();
    v16bf a0 = load_frag_a(&As[(wr * 32 +  0) * BKP]);
    v16bf a1 = load_frag_a(&As[(wr * 32 + 16) * BKP]);
#pragma unroll
    for (int j = 0; j < 4; ++j) {
      v16bf bj = load_frag_b(&Bs[(wc * 64 + j * 16) * BKP]);
      acc0[j] = WMMA_BF16(a0, bj, acc0[j]);
      acc1[j] = WMMA_BF16(a1, bj, acc1[j]);
    }
  }

  const int col = lane & 15, h = lane >> 4;
#pragma unroll
  for (int s = 0; s < 2; ++s)
#pragma unroll
    for (int j = 0; j < 4; ++j) {
      v8f c = s ? acc1[j] : acc0[j];
#pragma unroll
      for (int v = 0; v < 8; ++v) {
        const int nrow = n0 + wr * 32 + s * 16 + v + 8 * h;
        const int mcol = m0 + wc * 64 + j * 16 + col;
        Cb[(size_t)nrow * M + mcol] = (__bf16)c[v];
      }
    }
}

// ---------------------------------------------------------------------------
// y[b,t,n] = (sum_m qp[b,t,m] * kptv[b,n,m]) * dscale[b,t]   (f32 output)
// 64(t) x 128(n) block tile, 4 waves; wave = 32x64 = 2x4 frags.
// Operands are already bf16 in [row][k] layout -> pure-copy fills done with
// GLOBAL_LOAD_ASYNC_TO_LDS_B128 into ping-pong LDS buffers, overlapped with
// the WMMAs of the current tile (ASYNCcnt + barrier pipeline).
// ---------------------------------------------------------------------------
__global__ __launch_bounds__(128)
void ygemm_kernel(const __bf16* __restrict__ qp, const __bf16* __restrict__ kptvb,
                  const float* __restrict__ dscale, float* __restrict__ out) {
  using namespace cfg;
  __shared__ __align__(16) __bf16 As[2][64 * BKP];    // [t][m]
  __shared__ __align__(16) __bf16 Bs[2][128 * BKP];   // [n][m]
  const int b  = blockIdx.z;
  const int t0 = blockIdx.x * 64;
  const int n0 = blockIdx.y * 128;
  const __bf16* qpb = qp + (size_t)b * T * M;
  const __bf16* Cb  = kptvb + (size_t)b * DIM * M;
  const float*  dsb = dscale + (size_t)b * T;
  float*        ob  = out + (size_t)b * T * DIM;

  const int tid  = threadIdx.x;
  const int lane = tid & 31;
  const int wave = tid >> 5;
  const int wr = wave >> 1, wc = wave & 1;
  const int ra = tid >> 1;            // A fill: row 0..63
  const int ka = (tid & 1) * 16;      //         col 0 / 16

  auto fill_async = [&](int buf, int k0) {
    // A tile: 64 rows x 32 bf16  -> 2 x b128 per thread
    {
      uint32_t la = (uint32_t)(uintptr_t)&As[buf][ra * BKP + ka];
      const __bf16* ga = qpb + (size_t)(t0 + ra) * M + k0 + ka;
      async_copy_b128(la, ga);
      async_copy_b128(la + 16u, ga + 8);
    }
    // B tile: 128 rows x 32 bf16 -> 4 x b128 per thread (one row each)
    {
      uint32_t lb = (uint32_t)(uintptr_t)&Bs[buf][tid * BKP];
      const __bf16* gb = Cb + (size_t)(n0 + tid) * M + k0;
#pragma unroll
      for (int j = 0; j < 4; ++j) async_copy_b128(lb + 16u * j, gb + 8 * j);
    }
  };

  v8f acc0[4] = {}, acc1[4] = {};

  fill_async(0, 0);
  wait_async_zero();
  __syncthreads();

  for (int k0 = 0, it = 0; k0 < M; k0 += BK, ++it) {
    const int cur = it & 1;
    if (k0 + BK < M) fill_async(cur ^ 1, k0 + BK);   // overlap with WMMAs

    v16bf a0 = load_frag_a(&As[cur][(wr * 32 +  0) * BKP]);
    v16bf a1 = load_frag_a(&As[cur][(wr * 32 + 16) * BKP]);
#pragma unroll
    for (int j = 0; j < 4; ++j) {
      v16bf bj = load_frag_b(&Bs[cur][(wc * 64 + j * 16) * BKP]);
      acc0[j] = WMMA_BF16(a0, bj, acc0[j]);
      acc1[j] = WMMA_BF16(a1, bj, acc1[j]);
    }
    if (k0 + BK < M) {
      wait_async_zero();
      __syncthreads();
    }
  }

  const int col = lane & 15, h = lane >> 4;
#pragma unroll
  for (int s = 0; s < 2; ++s)
#pragma unroll
    for (int j = 0; j < 4; ++j) {
      v8f c = s ? acc1[j] : acc0[j];
#pragma unroll
      for (int v = 0; v < 8; ++v) {
        const int trow = t0 + wr * 32 + s * 16 + v + 8 * h;
        const int ncol = n0 + wc * 64 + j * 16 + col;
        ob[(size_t)trow * DIM + ncol] = c[v] * dsb[trow];
      }
    }
}

// ---------------------------------------------------------------------------
extern "C" void kernel_launch(void* const* d_in, const int* in_sizes, int n_in,
                              void* d_out, int out_size, void* d_ws, size_t ws_size,
                              hipStream_t stream) {
  using namespace cfg;
  const float* Q    = (const float*)d_in[0];
  const float* K    = (const float*)d_in[1];
  const float* V    = (const float*)d_in[2];
  const float* mask = (const float*)d_in[3];
  const float* w    = (const float*)d_in[4];
  float* out = (float*)d_out;

  // Workspace carve-up (~72 MB total)
  char* p = (char*)d_ws;
  auto take = [&](size_t bytes) {
    char* r = p;
    p += (bytes + 255) & ~(size_t)255;
    return (void*)r;
  };
  __bf16* qp    = (__bf16*)take((size_t)B * T * M * 2);
  __bf16* kp    = (__bf16*)take((size_t)B * T * M * 2);
  __bf16* Vb    = (__bf16*)take((size_t)B * T * DIM * 2);
  __bf16* kptvb = (__bf16*)take((size_t)B * DIM * M * 2);
  float*  kpsum = (float*)take((size_t)B * M * 4);
  float*  dsc   = (float*)take((size_t)B * T * 4);
  float*  xdQ   = (float*)take((size_t)B * T * 4);
  float*  xdK   = (float*)take((size_t)B * T * 4);

  const float rscale = 1.0f / sqrtf((float)M);

  rownorm_kernel<<<B * T / 8, 256, 0, stream>>>(Q, mask, xdQ, 0);
  rownorm_kernel<<<B * T / 8, 256, 0, stream>>>(K, mask, xdK, 1);
  convv_kernel<<<2048, 256, 0, stream>>>(V, mask, Vb);
  featurize_kernel<false><<<dim3(T / 64, M / 128, B), 128, 0, stream>>>(Q, w, xdQ, mask, qp, rscale);
  featurize_kernel<true><<<dim3(T / 64, M / 128, B), 128, 0, stream>>>(K, w, xdK, mask, kp, rscale);
  kpsum_kernel<<<dim3(M / 256, B), 256, 0, stream>>>(kp, kpsum);
  dscale_kernel<<<B * T / 8, 256, 0, stream>>>(qp, kpsum, dsc);
  kptv_kernel<<<dim3(DIM / 64, M / 128, B), 128, 0, stream>>>(Vb, kp, kptvb);
  ygemm_kernel<<<dim3(T / 64, DIM / 128, B), 128, 0, stream>>>(qp, kptvb, dsc, out);
}
